// BasicBlock_52742198395275
// MI455X (gfx1250) — matrix operand order repaired and checked by reading, not compiled, not run on hardware
//
#include <hip/hip_runtime.h>

#define IN   256
#define ANG  60
#define BAT  4
#define PI_F 3.14159265358979323846f

typedef __attribute__((ext_vector_type(16))) _Float16 v16h;
typedef __attribute__((ext_vector_type(8)))  _Float16 v8h;
typedef __attribute__((ext_vector_type(8)))  float    v8f;

// ---------------------------------------------------------------------------
// Bilinear sample with zero padding (matches jax map_coordinates order=1,
// mode='constant', cval=0)
// ---------------------------------------------------------------------------
__device__ __forceinline__ float bilin_img(const float* img, float sy, float sx) {
    float fy = floorf(sy), fx = floorf(sx);
    int y0 = (int)fy, x0 = (int)fx;
    int y1 = y0 + 1, x1 = x0 + 1;
    float wy = sy - fy, wx = sx - fx;
    float v00 = (y0 >= 0 && y0 < IN && x0 >= 0 && x0 < IN) ? img[y0 * IN + x0] : 0.f;
    float v01 = (y0 >= 0 && y0 < IN && x1 >= 0 && x1 < IN) ? img[y0 * IN + x1] : 0.f;
    float v10 = (y1 >= 0 && y1 < IN && x0 >= 0 && x0 < IN) ? img[y1 * IN + x0] : 0.f;
    float v11 = (y1 >= 0 && y1 < IN && x1 >= 0 && x1 < IN) ? img[y1 * IN + x1] : 0.f;
    return v00 * (1.f - wy) * (1.f - wx) + v01 * (1.f - wy) * wx +
           v10 * wy * (1.f - wx) + v11 * wy * wx;
}

// ---------------------------------------------------------------------------
// Radon forward + residual: resid[b,a,x] = sum_y rotate(img, -t)[y,x] - sino
// ---------------------------------------------------------------------------
__global__ void k_radon_resid(const float* __restrict__ x,
                              const float* __restrict__ theta,
                              const float* __restrict__ sino,
                              float* __restrict__ resid) {
    int tid = blockIdx.x * blockDim.x + threadIdx.x;
    if (tid >= BAT * ANG * IN) return;
    int xi = tid % IN;
    int a  = (tid / IN) % ANG;
    int b  = tid / (IN * ANG);
    const float* img = x + b * IN * IN;
    float t  = theta[a] * (PI_F / 180.f);
    float cs = cosf(t);            // cos(-t)
    float sn = -sinf(t);           // sin(-t)
    float c  = (IN - 1) * 0.5f;
    float xs = (float)xi - c;
    float acc = 0.f;
    for (int y = 0; y < IN; ++y) {
        float ys = (float)y - c;
        float sx =  cs * xs + sn * ys + c;
        float sy = -sn * xs + cs * ys + c;
        acc += bilin_img(img, sy, sx);
    }
    resid[tid] = acc - sino[tid];
}

// ---------------------------------------------------------------------------
// Unfiltered backprojection + gradient step.
// ---------------------------------------------------------------------------
__global__ void k_backproj(const float* __restrict__ x,
                           const float* __restrict__ theta,
                           const float* __restrict__ resid,
                           const float* __restrict__ lambda_step,
                           float* __restrict__ x_input) {
    int tid = blockIdx.x * blockDim.x + threadIdx.x;
    if (tid >= BAT * IN * IN) return;
    int xi = tid % IN;
    int yi = (tid / IN) % IN;
    int b  = tid / (IN * IN);
    float c  = (IN - 1) * 0.5f;
    float xs = (float)xi - c;
    float ys = (float)yi - c;
    float acc = 0.f;
    for (int a = 0; a < ANG; ++a) {
        float t  = theta[a] * (PI_F / 180.f);
        float cs = cosf(t), sn = sinf(t);
        float sx =  cs * xs + sn * ys + c;
        float sy = -sn * xs + cs * ys + c;
        const float* row = resid + (b * ANG + a) * IN;
        float fy = floorf(sy), fx = floorf(sx);
        int y0 = (int)fy, x0 = (int)fx;
        int y1 = y0 + 1, x1 = x0 + 1;
        float wy = sy - fy, wx = sx - fx;
        float v00 = (y0 >= 0 && y0 < IN && x0 >= 0 && x0 < IN) ? row[x0] : 0.f;
        float v01 = (y0 >= 0 && y0 < IN && x1 >= 0 && x1 < IN) ? row[x1] : 0.f;
        float v10 = (y1 >= 0 && y1 < IN && x0 >= 0 && x0 < IN) ? row[x0] : 0.f;
        float v11 = (y1 >= 0 && y1 < IN && x1 >= 0 && x1 < IN) ? row[x1] : 0.f;
        acc += v00 * (1.f - wy) * (1.f - wx) + v01 * (1.f - wy) * wx +
               v10 * wy * (1.f - wx) + v11 * wy * wx;
    }
    float scale = PI_F / (2.f * (float)ANG);
    x_input[tid] = x[tid] - lambda_step[0] * scale * acc;
}

// ---------------------------------------------------------------------------
// conv_D: 1 -> 32 channels, direct (K=9, too small for WMMA). fp16 out.
// ---------------------------------------------------------------------------
__global__ void k_convD(const float* __restrict__ xin,
                        const float* __restrict__ w,   // [32][1][3][3]
                        _Float16* __restrict__ out) {
    int tid = blockIdx.x * blockDim.x + threadIdx.x;
    if (tid >= BAT * 32 * IN * IN) return;
    int xi = tid % IN;
    int yi = (tid / IN) % IN;
    int co = (tid / (IN * IN)) % 32;
    int b  = tid / (IN * IN * 32);
    const float* img = xin + b * IN * IN;
    float acc = 0.f;
    #pragma unroll
    for (int ky = 0; ky < 3; ++ky) {
        int gy = yi + ky - 1;
        if (gy < 0 || gy >= IN) continue;
        #pragma unroll
        for (int kx = 0; kx < 3; ++kx) {
            int gx = xi + kx - 1;
            if (gx < 0 || gx >= IN) continue;
            acc += img[gy * IN + gx] * w[(co * 3 + ky) * 3 + kx];
        }
    }
    out[tid] = (_Float16)acc;
}

// ---------------------------------------------------------------------------
// Repack 32x32x3x3 fp32 weights (OIHW) into fp16 wp[set][tap][cout][cin]:
// each lane's B fragment (fixed cout, K=cin run of 16) is then one contiguous
// 32-byte load.
// ---------------------------------------------------------------------------
__global__ void k_pack(const float* __restrict__ w1f, const float* __restrict__ w2f,
                       const float* __restrict__ w1b, const float* __restrict__ w2b,
                       _Float16* __restrict__ wp) {
    int tid = blockIdx.x * blockDim.x + threadIdx.x;
    if (tid >= 4 * 9216) return;
    int set = tid / 9216;
    int r   = tid % 9216;
    int tap = r / 1024;
    int rem = r % 1024;
    int co  = rem / 32;       // [tap][cout][cin]
    int ci  = rem % 32;
    int ky  = tap / 3, kx = tap % 3;
    const float* src = (set == 0) ? w1f : (set == 1) ? w2f : (set == 2) ? w1b : w2b;
    wp[tid] = (_Float16)src[((co * 32 + ci) * 3 + ky) * 3 + kx];
}

// ---------------------------------------------------------------------------
// 32->32 3x3 SAME conv as implicit GEMM on WMMA.
//   M = 16 pixels/wave, N = 32 couts (2 x 16x16 tiles), K = 9 taps x 32 cin.
// Block = 128 threads (4 waves) covering 64 consecutive x of one image row.
// LDS patch is PIXEL-MAJOR [ky][x][cin] so each lane's A fragment is two
// contiguous 16-byte chunks -> 2x ds_load_b128 per tap. Weights are packed
// [tap][cout][cin] so each B fragment is one contiguous 32-byte global load.
// D stores: each lane owns 8 consecutive x at fixed cout -> global_store_b128.
// ---------------------------------------------------------------------------
template <int RELU>
__global__ void __launch_bounds__(128)
k_conv3x3_wmma(const _Float16* __restrict__ in,
               const _Float16* __restrict__ wp,   // [9][32][32] fp16
               _Float16* __restrict__ out) {
    __shared__ __align__(16) _Float16 lds_in[3 * 66 * 32];  // [ky][x][cin]

    int bid = blockIdx.x;
    int tx  = bid & 3;          // 64-px x tile
    int y   = (bid >> 2) & 255; // image row
    int b   = bid >> 10;        // batch
    int tid = threadIdx.x;
    int x0t = tx * 64;

    __builtin_prefetch(wp, 0, 3);   // global_prefetch_b8 on the weight block

    // Stage halo'd patch, transposing [cin][x] -> [x][cin] on the fly.
    // Global reads coalesced along x; LDS writes strided by 32 halves.
    for (int i = tid; i < 3 * 32 * 66; i += 128) {
        int xx = i % 66;
        int ci = (i / 66) % 32;
        int ky = i / (66 * 32);
        int gy = y + ky - 1;
        int gx = x0t + xx - 1;
        _Float16 v = (_Float16)0.f;
        if (gy >= 0 && gy < IN && gx >= 0 && gx < IN)
            v = in[((b * 32 + ci) * IN + gy) * IN + gx];
        lds_in[(ky * 66 + xx) * 32 + ci] = v;
    }
    __syncthreads();

    int wave = tid >> 5;
    int lane = tid & 31;
    int ml   = lane & 15;       // M row (A) / N col (B,D)
    int hi   = lane >> 4;       // lane half: selects K sub-range
    int px   = wave * 16 + ml;  // pixel owned by this lane's A row

    v8f acc0 = {};              // couts 0..15
    v8f acc1 = {};              // couts 16..31

    #pragma unroll
    for (int tap = 0; tap < 9; ++tap) {
        const int ky = tap / 3, kx = tap % 3;
        // A fragment: lane needs cin run hi*16..hi*16+15
        // at pixel px+kx -> two contiguous 16B LDS chunks.
        const _Float16* ap = &lds_in[(ky * 66 + px + kx) * 32];
        v8h alo = *(const v8h*)(ap + hi * 16);
        v8h ahi = *(const v8h*)(ap + hi * 16 + 8);
        v16h a  = __builtin_shufflevector(alo, ahi, 0, 1, 2, 3, 4, 5, 6, 7,
                                          8, 9, 10, 11, 12, 13, 14, 15);
        // B fragments: lane needs cin run hi*16..hi*16+15 at fixed cout
        // -> one contiguous 32B global load each.
        const _Float16* wt = wp + tap * 1024;
        v16h b0 = *(const v16h*)(wt + ml * 32 + hi * 16);         // cout = ml
        v16h b1 = *(const v16h*)(wt + (16 + ml) * 32 + hi * 16);  // cout = 16+ml

        acc0 = __builtin_amdgcn_wmma_f32_16x16x32_f16(false, a, false, b0,
                                                      (short)0, acc0, false, false);
        acc1 = __builtin_amdgcn_wmma_f32_16x16x32_f16(false, a, false, b1,
                                                      (short)0, acc1, false, false);
    }

    // D layout: VGPR r -> M = r + 8*hi, N = ml. Each lane owns 8 consecutive
    // x positions at fixed cout -> one 16B store per accumulator.
    int xbase = x0t + wave * 16 + hi * 8;
    v8h h0, h1;
    #pragma unroll
    for (int r = 0; r < 8; ++r) {
        float v0 = acc0[r], v1 = acc1[r];
        if (RELU) { v0 = fmaxf(v0, 0.f); v1 = fmaxf(v1, 0.f); }
        h0[r] = (_Float16)v0;
        h1[r] = (_Float16)v1;
    }
    *(v8h*)(out + ((b * 32 + ml) * IN + y) * IN + xbase)      = h0;
    *(v8h*)(out + ((b * 32 + 16 + ml) * IN + y) * IN + xbase) = h1;
}

// ---------------------------------------------------------------------------
// Soft threshold: h = sign(v) * relu(|v| - thr)
// ---------------------------------------------------------------------------
__global__ void k_soft(const _Float16* __restrict__ xf,
                       const float* __restrict__ thr_p,
                       _Float16* __restrict__ out) {
    int tid = blockIdx.x * blockDim.x + threadIdx.x;
    if (tid * 8 >= BAT * 32 * IN * IN) return;
    float t = thr_p[0];
    v8h v = *(const v8h*)(xf + tid * 8);
    v8h o;
    #pragma unroll
    for (int j = 0; j < 8; ++j) {
        float f = (float)v[j];
        o[j] = (_Float16)copysignf(fmaxf(fabsf(f) - t, 0.f), f);
    }
    *(v8h*)(out + tid * 8) = o;
}

// ---------------------------------------------------------------------------
// conv_G (32 -> 1) fused with residual add: x_pred = x_input + conv_G(x_bwd)
// ---------------------------------------------------------------------------
__global__ void k_convG_add(const _Float16* __restrict__ xb,
                            const float* __restrict__ wg,   // [1][32][3][3]
                            const float* __restrict__ x_input,
                            float* __restrict__ out) {
    int tid = blockIdx.x * blockDim.x + threadIdx.x;
    if (tid >= BAT * IN * IN) return;
    int xi = tid % IN;
    int yi = (tid / IN) % IN;
    int b  = tid / (IN * IN);
    float acc = 0.f;
    for (int ci = 0; ci < 32; ++ci) {
        const _Float16* plane = xb + ((b * 32 + ci) * IN) * IN;
        #pragma unroll
        for (int ky = 0; ky < 3; ++ky) {
            int gy = yi + ky - 1;
            if (gy < 0 || gy >= IN) continue;
            #pragma unroll
            for (int kx = 0; kx < 3; ++kx) {
                int gx = xi + kx - 1;
                if (gx < 0 || gx >= IN) continue;
                acc += (float)plane[gy * IN + gx] * wg[(ci * 3 + ky) * 3 + kx];
            }
        }
    }
    out[tid] = x_input[tid] + acc;
}

// ---------------------------------------------------------------------------
// symloss = x_D_est - x_D  (fp16 operands -> fp32 output), 8-wide
// ---------------------------------------------------------------------------
__global__ void k_symloss(const _Float16* __restrict__ est,
                          const _Float16* __restrict__ xD,
                          float* __restrict__ out) {
    int tid = blockIdx.x * blockDim.x + threadIdx.x;
    if (tid * 8 >= BAT * 32 * IN * IN) return;
    v8h e = *(const v8h*)(est + tid * 8);
    v8h d = *(const v8h*)(xD + tid * 8);
    #pragma unroll
    for (int j = 0; j < 8; ++j)
        out[tid * 8 + j] = (float)e[j] - (float)d[j];
}

// ---------------------------------------------------------------------------
// Host-side orchestration (graph-capture safe: only kernel launches).
// ---------------------------------------------------------------------------
extern "C" void kernel_launch(void* const* d_in, const int* in_sizes, int n_in,
                              void* d_out, int out_size, void* d_ws, size_t ws_size,
                              hipStream_t stream) {
    const float* x     = (const float*)d_in[0];
    const float* theta = (const float*)d_in[1];
    const float* sino  = (const float*)d_in[2];
    const float* lam   = (const float*)d_in[3];
    const float* thr   = (const float*)d_in[4];
    const float* wD    = (const float*)d_in[5];
    const float* w1f   = (const float*)d_in[6];
    const float* w2f   = (const float*)d_in[7];
    const float* w1b   = (const float*)d_in[8];
    const float* w2b   = (const float*)d_in[9];
    const float* wG    = (const float*)d_in[10];

    float* out = (float*)d_out;
    float* out_pred = out;                       // [4,1,256,256]
    float* out_sym  = out + BAT * IN * IN;       // [4,32,256,256]

    // Workspace layout (bytes)
    char* ws = (char*)d_ws;
    const size_t ACT = (size_t)BAT * 32 * IN * IN * sizeof(_Float16); // 16 MiB
    float*    resid  = (float*)(ws + 0);                    //  240 KiB
    float*    x_inp  = (float*)(ws + (1u << 18));           //    1 MiB
    _Float16* wpack  = (_Float16*)(ws + (5u << 18));        //   72 KiB
    _Float16* t0     = (_Float16*)(ws + (6u << 18));        // x_D
    _Float16* t1     = (_Float16*)((char*)t0 + ACT);        // scratch
    _Float16* t2     = (_Float16*)((char*)t1 + ACT);        // x_forward
    _Float16* t3     = (_Float16*)((char*)t2 + ACT);        // scratch
    _Float16* wp1f = wpack;
    _Float16* wp2f = wpack + 9216;
    _Float16* wp1b = wpack + 2 * 9216;
    _Float16* wp2b = wpack + 3 * 9216;

    const int TPB = 256;
    const int nPix  = BAT * IN * IN;          // 262144
    const int nAct  = BAT * 32 * IN * IN;     // 8388608
    const int nSino = BAT * ANG * IN;         // 61440

    // weight packing + data-consistency gradient step
    k_pack<<<(4 * 9216 + TPB - 1) / TPB, TPB, 0, stream>>>(w1f, w2f, w1b, w2b, wpack);
    k_radon_resid<<<(nSino + TPB - 1) / TPB, TPB, 0, stream>>>(x, theta, sino, resid);
    k_backproj<<<(nPix + TPB - 1) / TPB, TPB, 0, stream>>>(x, theta, resid, lam, x_inp);

    // sparse-coding stack
    k_convD<<<(nAct + TPB - 1) / TPB, TPB, 0, stream>>>(x_inp, wD, t0);   // x_D

    const int convBlocks = BAT * IN * (IN / 64);  // 4096
    k_conv3x3_wmma<1><<<convBlocks, 128, 0, stream>>>(t0, wp1f, t1);      // relu(conv1_f)
    k_conv3x3_wmma<0><<<convBlocks, 128, 0, stream>>>(t1, wp2f, t2);      // x_forward
    k_soft<<<(nAct / 8 + TPB - 1) / TPB, TPB, 0, stream>>>(t2, thr, t1);  // soft thr
    k_conv3x3_wmma<1><<<convBlocks, 128, 0, stream>>>(t1, wp1b, t3);      // relu(conv1_b)
    k_conv3x3_wmma<0><<<convBlocks, 128, 0, stream>>>(t3, wp2b, t1);      // x_backward
    k_convG_add<<<(nPix + TPB - 1) / TPB, TPB, 0, stream>>>(t1, wG, x_inp, out_pred);

    // symmetric-loss branch (decoder on un-thresholded code)
    k_conv3x3_wmma<1><<<convBlocks, 128, 0, stream>>>(t2, wp1b, t3);
    k_conv3x3_wmma<0><<<convBlocks, 128, 0, stream>>>(t3, wp2b, t1);      // x_D_est
    k_symloss<<<(nAct / 8 + TPB - 1) / TPB, TPB, 0, stream>>>(t1, t0, out_sym);
}
